// DGCNN_seg_63419487093385
// MI455X (gfx1250) — matrix operand order repaired and checked by reading, compiled, not run on hardware
//
#include <hip/hip_runtime.h>

#define KNN 40
#define CHUNK 512

typedef __attribute__((ext_vector_type(16))) __bf16    v16bf;
typedef __attribute__((ext_vector_type(16))) _Float16  v16h;
typedef __attribute__((ext_vector_type(8)))  float     v8f;

union BF16x16 { v16bf v; unsigned short u[16]; uint4 q[2]; };
union HF16x16 { v16h  v; unsigned short u[16]; uint4 q[2]; };

__device__ __forceinline__ unsigned short f2bfu(float f) {
  unsigned u = __builtin_bit_cast(unsigned, f);
  unsigned r = u + 0x7FFFu + ((u >> 16) & 1u);   // round-to-nearest-even
  return (unsigned short)(r >> 16);
}

__device__ __forceinline__ unsigned short f2hu(float f) {
  _Float16 h = (_Float16)f;
  return __builtin_bit_cast(unsigned short, h);
}

__device__ __forceinline__ v8f vzero8() {
  v8f z;
#pragma unroll
  for (int i = 0; i < 8; ++i) z[i] = 0.0f;
  return z;
}

__device__ __forceinline__ v8f wmma_bf16(v16bf a, v16bf b, v8f c) {
  return __builtin_amdgcn_wmma_f32_16x16x32_bf16(false, a, false, b, (short)0, c,
                                                 false, false);
}
__device__ __forceinline__ v8f wmma_f16(v16h a, v16h b, v8f c) {
  return __builtin_amdgcn_wmma_f32_16x16x32_f16(false, a, false, b, (short)0, c,
                                                false, false);
}

// A fragment (16x32, M x K) from LDS row-major [rows, ld] (16-bit elements)
// lane&15 = M row; element e -> k = (lane>>4)*16 + e
__device__ __forceinline__ v16bf a_frag_lds(const unsigned short* s, int ld,
                                            int rowbase, int kk, int lane) {
  BF16x16 f;
  const int r  = rowbase + (lane & 15);
  const int kb = kk + ((lane >> 4) << 4);
#pragma unroll
  for (int e = 0; e < 16; ++e)
    f.u[e] = s[r * ld + kb + e];
  return f.v;
}
__device__ __forceinline__ v16h a_frag_lds_h(const unsigned short* s, int ld,
                                             int rowbase, int kk, int lane) {
  HF16x16 f;
  const int r  = rowbase + (lane & 15);
  const int kb = kk + ((lane >> 4) << 4);
#pragma unroll
  for (int e = 0; e < 16; ++e)
    f.u[e] = s[r * ld + kb + e];
  return f.v;
}

// B fragment (32x16, K x N) from pre-packed weights.
// Packed layout: [ktile][ntile][lane][e] ushort, 32B per lane -> 2x b128 loads.
__device__ __forceinline__ v16bf b_frag_packed(const unsigned short* __restrict__ P,
                                               int ncTiles, int ktile, int ntile,
                                               int lane) {
  const uint4* q =
      (const uint4*)(P + ((((long)ktile * ncTiles + ntile) << 5) + lane) * 16);
  BF16x16 f;
  f.q[0] = q[0];
  f.q[1] = q[1];
  return f.v;
}

__device__ __forceinline__ float leakyf(float v) { return v > 0.f ? v : 0.2f * v; }

// ------------------------------------------------ weight pre-pack (f32 -> bf16)
__global__ __launch_bounds__(256) void packW_kernel(const float* __restrict__ W,
                                                    unsigned short* __restrict__ P,
                                                    int Kd, int Nc) {
  const int Kp = (Kd + 31) & ~31;
  const long total = (long)Kp * Nc;
  const long g = (long)blockIdx.x * 256 + threadIdx.x;
  if (g >= total) return;
  const int e = (int)(g & 15);
  const int lane = (int)((g >> 4) & 31);
  const long tile = g >> 9;
  const int ncT = Nc >> 4;
  const int ntile = (int)(tile % ncT);
  const int ktile = (int)(tile / ncT);
  const int k = ktile * 32 + ((lane >> 4) << 4) + e;
  const int n = ntile * 16 + (lane & 15);
  P[g] = (k < Kd) ? f2bfu(W[(long)k * Nc + n]) : (unsigned short)0;
}

// --------------------------------------------------------- squared norms |x|^2
__global__ __launch_bounds__(256) void sqnorm_kernel(const float* __restrict__ x,
                                                     float* __restrict__ sqn,
                                                     int total, int C) {
  const int g = blockIdx.x * blockDim.x + threadIdx.x;
  if (g >= total) return;
  float s = 0.f;
  for (int c = 0; c < C; ++c) { const float v = x[(long)g * C + c]; s += v * v; }
  sqn[g] = s;
}

// --------------------------------------------- KNN via WMMA Gram (16 queries)
// d2(q,m) = |q|^2 + |m|^2 - 2 q.m ; Gram on the f16 matrix pipe, f32 accum.
// Compile-time C: static aQ fragments (no movrels), unconditional B loads.
template <int CT>
__global__ __launch_bounds__(256) void knn_wmma_kernel(
    const float* __restrict__ x, const float* __restrict__ sqn,
    int* __restrict__ idx, int B, int N) {
  constexpr int Cp  = (CT + 31) & ~31;
  constexpr int NKT = Cp >> 5;                   // 1 (C<=32) or 4 (C=128)
  __shared__ unsigned short sQ[16 * Cp];         // query tile, f16
  __shared__ float sD[16 * (CHUNK + 4)];         // distance chunk, row-padded
  __shared__ float sqq[16];
  const int ldD = CHUNK + 4;
  const int tid = threadIdx.x, lane = tid & 31, wave = tid >> 5;
  const int b  = blockIdx.x / (N >> 4);
  const int q0 = (blockIdx.x % (N >> 4)) << 4;
  const float* xb = x + (long)b * N * CT;

  for (int e = tid; e < 16 * Cp; e += 256) {
    const int r = e / Cp, c = e - r * Cp;
    sQ[e] = (c < CT) ? f2hu(xb[(long)(q0 + r) * CT + c]) : (unsigned short)0;
  }
  if (tid < 16) sqq[tid] = sqn[(long)b * N + q0 + tid];
  __syncthreads();

  // query A fragments: built once, reused for all candidate tiles (static!)
  v16h aQ[NKT];
#pragma unroll
  for (int kt = 0; kt < NKT; ++kt) aQ[kt] = a_frag_lds_h(sQ, Cp, 0, kt * 32, lane);

  float bd[KNN]; int bi[KNN];
#pragma unroll
  for (int i = 0; i < KNN; ++i) { bd[i] = 3.4e38f; bi[i] = 0; }

  for (int ch = 0; ch < N; ch += CHUNK) {
#pragma unroll
    for (int t = 0; t < 4; ++t) {                // wave-uniform candidate tiles
      const int moff = (wave * 4 + t) << 4;
      const int m0 = ch + moff;
      v8f acc = vzero8();
      const long rowm = (long)(m0 + (lane & 15)) * CT;
#pragma unroll
      for (int kt = 0; kt < NKT; ++kt) {
        HF16x16 bf;                              // B: x[m0+lane&15][k] (f16)
        const int kb = kt * 32 + ((lane >> 4) << 4);
        if (CT == 128) {                         // k always valid, contiguous
#pragma unroll
          for (int e = 0; e < 16; ++e) bf.u[e] = f2hu(xb[rowm + kb + e]);
        } else if (CT == 16) {                   // valid iff lane<16 (kb==0)
          const bool lo = (lane < 16);
#pragma unroll
          for (int e = 0; e < 16; ++e) {
            const float v = xb[rowm + e];
            bf.u[e] = lo ? f2hu(v) : (unsigned short)0;
          }
        } else {                                 // CT==3: e<3 && lane<16
          const bool lo = (lane < 16);
#pragma unroll
          for (int e = 0; e < 16; ++e) {
            if (e < CT) {
              const float v = xb[rowm + e];
              bf.u[e] = lo ? f2hu(v) : (unsigned short)0;
            } else bf.u[e] = 0;
          }
        }
        acc = wmma_f16(aQ[kt], bf.v, acc);
      }
      const float sm = sqn[(long)b * N + m0 + (lane & 15)];
#pragma unroll
      for (int r8 = 0; r8 < 8; ++r8) {
        const int q = ((lane >> 4) << 3) + r8;
        sD[q * ldD + moff + (lane & 15)] = sqq[q] + sm - 2.f * acc[r8];
      }
    }
    __syncthreads();
    if (tid < 16) {                              // selector: merge chunk
      const float* row = sD + tid * ldD;
      for (int j = 0; j < CHUNK; ++j) {
        const float d = row[j];
        if (d < bd[KNN - 1]) {                   // stable: ties keep earlier m
          int jj = KNN - 1;
          while (jj > 0 && bd[jj - 1] > d) {
            bd[jj] = bd[jj - 1]; bi[jj] = bi[jj - 1]; --jj;
          }
          bd[jj] = d; bi[jj] = ch + j;
        }
      }
    }
    __syncthreads();
  }
  if (tid < 16) {
    int* o = idx + ((long)b * N + q0 + tid) * KNN;
    for (int i = 0; i < KNN; ++i) o[i] = bi[i];
  }
}

// ------------------------------------------------- EdgeConv (2 points / block)
__global__ __launch_bounds__(256) void edgeconv_kernel(
    const float* __restrict__ x, const int* __restrict__ idx,
    const unsigned short* __restrict__ PW1, const float* __restrict__ b1,
    const unsigned short* __restrict__ PW2, const float* __restrict__ b2,
    float* __restrict__ out, int B, int N, int Cin, int H, int O) {
  __shared__ unsigned short sA[80 * 256];   // edge features, bf16
  __shared__ unsigned short sH[80 * 128];   // hidden, bf16
  const int tid = threadIdx.x, lane = tid & 31, wave = tid >> 5;
  const int C2 = 2 * Cin;
  const int ldA = (C2 + 31) & ~31;          // == packed Kp of W1
  const int halfN = N >> 1;
  const int b  = blockIdx.x / halfN;
  const int p0 = (blockIdx.x % halfN) * 2;

  // Phase A: each wave stages 10 edge rows; idx loaded once per row,
  // channel traffic coalesced across lanes.
  for (int rr = 0; rr < 10; ++rr) {
    const int r  = wave * 10 + rr;
    const int pp = r / KNN, kk = r - pp * KNN;
    const int n  = p0 + pp;
    const int nb = idx[((long)b * N + n) * KNN + kk];   // wave-uniform
    const float* ctr = x + ((long)b * N + n)  * Cin;
    const float* nbr = x + ((long)b * N + nb) * Cin;
    for (int c = lane; c < Cin; c += 32) {
      const float cv = ctr[c];
      sA[r * ldA + c]       = f2bfu(nbr[c] - cv);
      sA[r * ldA + Cin + c] = f2bfu(cv);
    }
    for (int c = C2 + lane; c < ldA; c += 32) sA[r * ldA + c] = 0;  // pad
  }
  __syncthreads();

  const int hT = H >> 4;
  if (wave < hT) {
    const int colb = wave << 4;
    v8f acc[5];
#pragma unroll
    for (int t = 0; t < 5; ++t) acc[t] = vzero8();
    for (int kt = 0; kt < (ldA >> 5); ++kt) {
      const v16bf bf = b_frag_packed(PW1, hT, kt, wave, lane);
#pragma unroll
      for (int t = 0; t < 5; ++t)
        acc[t] = wmma_bf16(a_frag_lds(sA, ldA, t * 16, kt * 32, lane), bf, acc[t]);
    }
    const float bias = b1[colb + (lane & 15)];
#pragma unroll
    for (int t = 0; t < 5; ++t)
#pragma unroll
      for (int r8 = 0; r8 < 8; ++r8) {
        const int row = t * 16 + ((lane >> 4) << 3) + r8;
        sH[row * H + colb + (lane & 15)] = f2bfu(leakyf(acc[t][r8] + bias));
      }
  }
  __syncthreads();

  const int oT = O >> 4;
  if (wave < oT) {
    const int colb = wave << 4;
    v8f acc[5];
#pragma unroll
    for (int t = 0; t < 5; ++t) acc[t] = vzero8();
    for (int kt = 0; kt < (H >> 5); ++kt) {
      const v16bf bf = b_frag_packed(PW2, oT, kt, wave, lane);
#pragma unroll
      for (int t = 0; t < 5; ++t)
        acc[t] = wmma_bf16(a_frag_lds(sH, H, t * 16, kt * 32, lane), bf, acc[t]);
    }
    const float bias = b2[colb + (lane & 15)];
    float mx0 = -3.4e38f, mx1 = -3.4e38f;
#pragma unroll
    for (int t = 0; t < 5; ++t)
#pragma unroll
      for (int r8 = 0; r8 < 8; ++r8) {
        const int row = t * 16 + ((lane >> 4) << 3) + r8;
        const float v = leakyf(acc[t][r8] + bias);
        if (row < KNN) mx0 = fmaxf(mx0, v); else mx1 = fmaxf(mx1, v);
      }
    mx0 = fmaxf(mx0, __shfl_xor(mx0, 16, 32));   // combine lane halves (rows)
    mx1 = fmaxf(mx1, __shfl_xor(mx1, 16, 32));
    if (lane < 16) {
      out[((long)b * N + p0)     * O + colb + lane] = mx0;
      out[((long)b * N + p0 + 1) * O + colb + lane] = mx1;
    }
  }
}

// ----------------------------------- generic GEMM: out = act(A@W+b) [+ resid]
__global__ __launch_bounds__(256) void gemm_kernel(
    const float* __restrict__ A, const unsigned short* __restrict__ PW,
    const float* __restrict__ bias, const float* __restrict__ resid,
    float* __restrict__ out, int M, int Kd, int Nc, int act) {
  __shared__ unsigned short sA[16 * 1024];  // Kd <= 1024
  const int tid = threadIdx.x, lane = tid & 31, wave = tid >> 5;
  const int rowb = blockIdx.x * 16;
  const int ldA = (Kd + 31) & ~31;          // == packed Kp
  for (int e = tid; e < 16 * ldA; e += 256) {
    const int r = e / ldA, c = e - r * ldA;
    const int row = rowb + r;
    sA[e] = (c < Kd && row < M) ? f2bfu(A[(long)row * Kd + c]) : (unsigned short)0;
  }
  __syncthreads();
  const int colb = blockIdx.y * 128 + wave * 16;
  if (colb < Nc) {
    const int ncT = Nc >> 4, ntile = colb >> 4;
    v8f acc = vzero8();
    for (int kt = 0; kt < (ldA >> 5); ++kt)
      acc = wmma_bf16(a_frag_lds(sA, ldA, 0, kt * 32, lane),
                      b_frag_packed(PW, ncT, kt, ntile, lane), acc);
    const float bv = bias ? bias[colb + (lane & 15)] : 0.f;
#pragma unroll
    for (int r8 = 0; r8 < 8; ++r8) {
      const int row = rowb + ((lane >> 4) << 3) + r8;
      if (row < M) {
        float v = acc[r8] + bv;
        if (act == 1) v = fmaxf(v, 0.f);
        else if (act == 2) v = leakyf(v);
        if (resid) v += resid[(long)row * Nc + colb + (lane & 15)];
        out[(long)row * Nc + colb + (lane & 15)] = v;
      }
    }
  }
}

// -------- fused TNet stage 2: out[b,c] = max_n leaky(A[b,n,:128]@W[:,c]+b[c])
__global__ __launch_bounds__(256) void t2max_kernel(
    const float* __restrict__ A, const unsigned short* __restrict__ PW,
    const float* __restrict__ bias, float* __restrict__ out, int N) {
  __shared__ unsigned short sA[16 * 128];
  const int tid = threadIdx.x, lane = tid & 31, wave = tid >> 5;
  const int b = blockIdx.x;
  const int colb = blockIdx.y * 128 + wave * 16;
  const int ntile = colb >> 4;
  v16bf bfr[4];
#pragma unroll
  for (int kt = 0; kt < 4; ++kt)
    bfr[kt] = b_frag_packed(PW, 1024 >> 4, kt, ntile, lane);
  const float bv = bias[colb + (lane & 15)];
  float runmax = -3.4e38f;
  for (int n0 = 0; n0 < N; n0 += 16) {
    __syncthreads();
    for (int e = tid; e < 16 * 128; e += 256)
      sA[e] = f2bfu(A[((long)b * N + n0) * 128 + e]);
    __syncthreads();
    v8f acc = vzero8();
#pragma unroll
    for (int kt = 0; kt < 4; ++kt)
      acc = wmma_bf16(a_frag_lds(sA, 128, 0, kt * 32, lane), bfr[kt], acc);
#pragma unroll
    for (int r8 = 0; r8 < 8; ++r8)
      runmax = fmaxf(runmax, leakyf(acc[r8] + bv));
  }
  runmax = fmaxf(runmax, __shfl_xor(runmax, 16, 32));
  if (lane < 16) out[(long)b * 1024 + colb + lane] = runmax;
}

// ----------------------------------------------- tiny: t4 = t3b@W[256,9]+b[9]
__global__ void t4_kernel(const float* __restrict__ A, const float* __restrict__ W,
                          const float* __restrict__ bias, float* __restrict__ out,
                          int B) {
  const int g = blockIdx.x * blockDim.x + threadIdx.x;
  if (g >= B * 9) return;
  const int b = g / 9, j = g - b * 9;
  float s = bias[j];
  for (int k = 0; k < 256; ++k) s += A[b * 256 + k] * W[k * 9 + j];
  out[g] = s;
}

// ------------- x0 = pos @ T(3x3); trunk input = cat(x0, features) [B,N,16]
__global__ void tnet_apply_kernel(const float* __restrict__ pos,
                                  const float* __restrict__ feat,
                                  const float* __restrict__ tmat,
                                  float* __restrict__ out, int B, int N) {
  const int g = blockIdx.x * blockDim.x + threadIdx.x;
  if (g >= B * N) return;
  const int b = g / N;
  const float* p = pos + (long)g * 3;
  const float* T = tmat + b * 9;
  float* o = out + (long)g * 16;
  o[0] = p[0] * T[0] + p[1] * T[3] + p[2] * T[6];
  o[1] = p[0] * T[1] + p[1] * T[4] + p[2] * T[7];
  o[2] = p[0] * T[2] + p[1] * T[5] + p[2] * T[8];
  const float* f = feat + (long)g * 13;
#pragma unroll
  for (int c = 0; c < 13; ++c) o[3 + c] = f[c];
}

// =============================================================================
extern "C" void kernel_launch(void* const* d_in, const int* in_sizes, int n_in,
                              void* d_out, int out_size, void* d_ws, size_t ws_size,
                              hipStream_t stream) {
  (void)in_sizes; (void)n_in; (void)out_size; (void)ws_size;
  const int B = 4, N = 2048, M = B * N;
  const float* pos  = (const float*)d_in[0];
  const float* feat = (const float*)d_in[1];
  // params flattened in sorted-dict-key leaf order: conv, lin, lt, t1, t2, t3, t4
  int p = 2;
  const float *convW1[3], *convB1[3], *convW2[3], *convB2[3];
  for (int i = 0; i < 3; ++i) {
    convW1[i] = (const float*)d_in[p++]; convB1[i] = (const float*)d_in[p++];
    convW2[i] = (const float*)d_in[p++]; convB2[i] = (const float*)d_in[p++];
  }
  const float *linW1[3], *linB1[3], *linW2[3], *linB2[3];
  for (int i = 0; i < 3; ++i) {
    linW1[i] = (const float*)d_in[p++]; linB1[i] = (const float*)d_in[p++];
    linW2[i] = (const float*)d_in[p++]; linB2[i] = (const float*)d_in[p++];
  }
  const float *ltW[3], *ltB[3];
  for (int i = 0; i < 3; ++i) {
    ltW[i] = (const float*)d_in[p++]; ltB[i] = (const float*)d_in[p++];
  }
  const float* t1W1 = (const float*)d_in[p++]; const float* t1B1 = (const float*)d_in[p++];
  const float* t1W2 = (const float*)d_in[p++]; const float* t1B2 = (const float*)d_in[p++];
  const float* t2W  = (const float*)d_in[p++]; const float* t2B  = (const float*)d_in[p++];
  const float* t3W1 = (const float*)d_in[p++]; const float* t3B1 = (const float*)d_in[p++];
  const float* t3W2 = (const float*)d_in[p++]; const float* t3B2 = (const float*)d_in[p++];
  const float* t4W  = (const float*)d_in[p++]; const float* t4B  = (const float*)d_in[p++];

  size_t off = 0;
  auto carve = [&](size_t bytes) -> void* {
    void* q = (char*)d_ws + off;
    off += (bytes + 255) & ~(size_t)255;
    return q;
  };
  int*   idxb = (int*)  carve((size_t)M * KNN * sizeof(int));
  float* sqnb = (float*)carve((size_t)M * sizeof(float));
  float* actA = (float*)carve((size_t)M * 128 * sizeof(float));
  float* actB = (float*)carve((size_t)M * 128 * sizeof(float));
  float* actC = (float*)carve((size_t)M * 128 * sizeof(float));
  float* tmax = (float*)carve((size_t)B * 1024 * sizeof(float));
  float* t3a  = (float*)carve((size_t)B * 512 * sizeof(float));
  float* t3b  = (float*)carve((size_t)B * 256 * sizeof(float));
  float* tmat = (float*)carve((size_t)B * 9 * sizeof(float));

  // packed bf16 weights (Kp x Nc ushorts each, Kp = roundup32)
  auto carveW = [&](int Kd, int Nc) -> unsigned short* {
    const int Kp = (Kd + 31) & ~31;
    return (unsigned short*)carve((size_t)Kp * Nc * sizeof(unsigned short));
  };
  unsigned short *pconvW1[3], *pconvW2[3], *plinW1[3], *plinW2[3], *pltW[3];
  const int dinL[3] = {16, 128, 128};
  for (int i = 0; i < 3; ++i) {
    pconvW1[i] = carveW(2 * dinL[i], 128);
    pconvW2[i] = carveW(128, 128);
    plinW1[i]  = carveW(128, 128);
    plinW2[i]  = carveW(128, 128);
    pltW[i]    = carveW(dinL[i], 128);
  }
  unsigned short* pt1W1 = carveW(6, 64);
  unsigned short* pt1W2 = carveW(64, 128);
  unsigned short* pt2W  = carveW(128, 1024);
  unsigned short* pt3W1 = carveW(1024, 512);
  unsigned short* pt3W2 = carveW(512, 256);

  dim3 blk(256);
  auto pack = [&](const float* W, unsigned short* P, int Kd, int Nc) {
    const long total = (long)((Kd + 31) & ~31) * Nc;
    packW_kernel<<<dim3((unsigned)((total + 255) / 256)), blk, 0, stream>>>(W, P, Kd, Nc);
  };
  for (int i = 0; i < 3; ++i) {
    pack(convW1[i], pconvW1[i], 2 * dinL[i], 128);
    pack(convW2[i], pconvW2[i], 128, 128);
    pack(linW1[i],  plinW1[i],  128, 128);
    pack(linW2[i],  plinW2[i],  128, 128);
    pack(ltW[i],    pltW[i],    dinL[i], 128);
  }
  pack(t1W1, pt1W1, 6, 64);
  pack(t1W2, pt1W2, 64, 128);
  pack(t2W,  pt2W,  128, 1024);
  pack(t3W1, pt3W1, 1024, 512);
  pack(t3W2, pt3W2, 512, 256);

  const int knnGrid = B * (N >> 4);
  auto knn = [&](const float* x, int C) {
    sqnorm_kernel<<<(M + 255) / 256, blk, 0, stream>>>(x, sqnb, M, C);
    if (C == 3)
      knn_wmma_kernel<3><<<knnGrid, blk, 0, stream>>>(x, sqnb, idxb, B, N);
    else if (C == 16)
      knn_wmma_kernel<16><<<knnGrid, blk, 0, stream>>>(x, sqnb, idxb, B, N);
    else
      knn_wmma_kernel<128><<<knnGrid, blk, 0, stream>>>(x, sqnb, idxb, B, N);
  };

  // ---------------- TNet ----------------
  knn(pos, 3);
  edgeconv_kernel<<<M / 2, blk, 0, stream>>>(pos, idxb, pt1W1, t1B1, pt1W2, t1B2,
                                             actA, B, N, 3, 64, 128);
  t2max_kernel<<<dim3(B, 8), blk, 0, stream>>>(actA, pt2W, t2B, tmax, N);
  gemm_kernel<<<dim3(1, 4), blk, 0, stream>>>(tmax, pt3W1, t3B1, nullptr, t3a,
                                              B, 1024, 512, 2);
  gemm_kernel<<<dim3(1, 2), blk, 0, stream>>>(t3a, pt3W2, t3B2, nullptr, t3b,
                                              B, 512, 256, 2);
  t4_kernel<<<1, 64, 0, stream>>>(t3b, t4W, t4B, tmat, B);
  tnet_apply_kernel<<<(M + 255) / 256, blk, 0, stream>>>(pos, feat, tmat, actB, B, N);

  // ---------------- trunk ----------------
  float* xcur = actB;
  for (int i = 0; i < 3; ++i) {
    const int din = dinL[i];
    float* xnext = (i == 0) ? actC : (i == 1) ? actB : (float*)d_out;
    knn(xcur, din);
    edgeconv_kernel<<<M / 2, blk, 0, stream>>>(xcur, idxb, pconvW1[i], convB1[i],
                                               pconvW2[i], convB2[i], actA,
                                               B, N, din, 128, 128);
    // xi = relu(xi@W1+b1)@W2+b2 ; x = x@Wt+bt + xi
    gemm_kernel<<<dim3(M / 16, 1), blk, 0, stream>>>(actA, plinW1[i], linB1[i],
                                                     nullptr, xnext, M, 128, 128, 1);
    gemm_kernel<<<dim3(M / 16, 1), blk, 0, stream>>>(xnext, plinW2[i], linB2[i],
                                                     nullptr, actA, M, 128, 128, 0);
    gemm_kernel<<<dim3(M / 16, 1), blk, 0, stream>>>(xcur, pltW[i], ltB[i],
                                                     actA, xnext, M, din, 128, 0);
    xcur = xnext;
  }
}
